// Nodes_Clustering_Learn_45286135169448
// MI455X (gfx1250) — matrix operand order repaired and checked by reading, not compiled, not run on hardware
//
#include <hip/hip_runtime.h>
#include <hip/hip_bf16.h>
#include <math.h>

typedef __attribute__((ext_vector_type(16))) _Float16 v16h;
typedef __attribute__((ext_vector_type(8)))  float    v8f;

#define BB 4
#define NN 512
#define DD 64
#define CC 16

// gfx1250 hardware V_TANH_F32.
#if defined(__has_builtin)
#if __has_builtin(__builtin_amdgcn_tanhf)
#define TANH_F32(x) __builtin_amdgcn_tanhf(x)
#endif
#endif
#ifndef TANH_F32
#define TANH_F32(x) tanhf(x)
#endif

// Exchange with lane^16 in one DS op (group-of-32 swizzle, SWAPX16):
// offset = and_mask 0x1f | or_mask 0<<5 | xor_mask 0x10<<10 = 0x401F
__device__ __forceinline__ float swap16(float x) {
    int xi = __builtin_bit_cast(int, x);
    xi = __builtin_amdgcn_ds_swizzle(xi, 0x401F);
    return __builtin_bit_cast(float, xi);
}

// ---------------------------------------------------------------------------
// Kernel 1: key2 / query2 precompute. One 64-thread block per (b,n) row.
// ---------------------------------------------------------------------------
__global__ __launch_bounds__(64) void ncl_linear2_kernel(
    const float* __restrict__ features,
    const float* __restrict__ Wk, const float* __restrict__ bk,
    const float* __restrict__ Wq, const float* __restrict__ bq,
    float* __restrict__ key2, float* __restrict__ query2)
{
    __shared__ float xs[DD];
    __shared__ float ts[DD];

    const int row = blockIdx.x;      // 0 .. B*N-1
    const int e   = threadIdx.x;     // output feature index 0..63

    xs[e] = features[(size_t)row * DD + e];
    __syncthreads();

    {   // key chain (applied twice, relu between)
        const float* w = Wk + e * DD;
        float acc = bk[e];
        #pragma unroll 16
        for (int d = 0; d < DD; ++d) acc += xs[d] * w[d];
        ts[e] = fmaxf(acc, 0.0f);
        __syncthreads();
        float acc2 = bk[e];
        #pragma unroll 16
        for (int d = 0; d < DD; ++d) acc2 += ts[d] * w[d];
        key2[(size_t)row * DD + e] = fmaxf(acc2, 0.0f);
    }
    __syncthreads();

    {   // query chain
        const float* w = Wq + e * DD;
        float acc = bq[e];
        #pragma unroll 16
        for (int d = 0; d < DD; ++d) acc += xs[d] * w[d];
        ts[e] = fmaxf(acc, 0.0f);
        __syncthreads();
        float acc2 = bq[e];
        #pragma unroll 16
        for (int d = 0; d < DD; ++d) acc2 += ts[d] * w[d];
        query2[(size_t)row * DD + e] = fmaxf(acc2, 0.0f);
    }
}

// ---------------------------------------------------------------------------
// Kernel 2: one wave32 per (b, i-tile(16), j-tile(16)), loops 16 j's.
//   A[c][k]   = Wc[c][k]                      (16 x 64, static per wave)
//   B[k][m]   = tanh(key2[i0+m][k] + query2[j][k])     (64 x 16)
//   D[c][m]   = A x B  via 2x v_wmma_f32_16x16x32_f16
// Lane l owns pair i = i0 + l%16, classes 8h..8h+7 (h = l>>4) in acc[0..7].
// Softmax over classes: 7 in-lane ops + one lane^16 swizzle (max and sum).
// Output: 8 consecutive floats per lane -> two b128 stores per j.
// ---------------------------------------------------------------------------
__global__ __launch_bounds__(256) void ncl_pair_softmax_kernel(
    const float* __restrict__ key2,    // [B,N,D]
    const float* __restrict__ query2,  // [B,N,D]
    const float* __restrict__ Wc,      // [C,D]
    const float* __restrict__ bc,      // [C]
    float* __restrict__ out)           // [B,N,N,C]
{
    const int lane = threadIdx.x & 31;
    const int wave = threadIdx.x >> 5;
    const int tile = blockIdx.x * 8 + wave;      // 0 .. 4095
    const int b  = tile >> 10;
    const int tr = tile & 1023;
    const int it = tr >> 5;
    const int jt = tr & 31;

    const int h = lane >> 4;     // 16-lane half
    const int m = lane & 15;     // pair index within tile / A row (class)

    const int i0 = it * 16;
    const int j0 = jt * 16;

    // ---- A fragment: Wc rows in contiguous k order (same order as B).
    //      chunk0: elements 0..15 -> K = h*16 + e
    //      chunk1: same + 32
    const float* wrow = Wc + m * DD;             // class c = m (A row)
    v16h af0, af1;
    #pragma unroll
    for (int e = 0; e < 8; ++e) {
        af0[e]     = (_Float16)wrow[h * 16 + e];
        af0[8 + e] = (_Float16)wrow[h * 16 + 8 + e];
        af1[e]     = (_Float16)wrow[32 + h * 16 + e];
        af1[8 + e] = (_Float16)wrow[40 + h * 16 + e];
    }

    // bias for the 8 classes this lane's acc VGPRs represent: c = 8h + r
    float bcr[8];
    #pragma unroll
    for (int r = 0; r < 8; ++r) bcr[r] = bc[8 * h + r];

    // ---- key-row values, hoisted out of the j loop.
    //      B chunk0 uses A's order K = h*16 + e (e=0..15), chunk1: +32  -> contiguous!
    const float* krow = key2 + ((size_t)b * NN + (i0 + m)) * DD;
    float kf0[16], kf1[16];
    #pragma unroll
    for (int e = 0; e < 16; ++e) {
        kf0[e] = krow[h * 16 + e];
        kf1[e] = krow[32 + h * 16 + e];
    }

    for (int jj = 0; jj < 16; ++jj) {
        const int j = j0 + jj;
        const float* qrow = query2 + ((size_t)b * NN + j) * DD;

        // ---- B fragments: tanh(key + query) in the same contiguous k order as A ----
        v16h bf0, bf1;
        #pragma unroll
        for (int e = 0; e < 16; ++e) {
            bf0[e] = (_Float16)TANH_F32(kf0[e] + qrow[h * 16 + e]);
            bf1[e] = (_Float16)TANH_F32(kf1[e] + qrow[32 + h * 16 + e]);
        }

        v8f acc = {};
        acc = __builtin_amdgcn_wmma_f32_16x16x32_f16(
            false, af0, false, bf0, (short)0, acc, false, false);
        acc = __builtin_amdgcn_wmma_f32_16x16x32_f16(
            false, af1, false, bf1, (short)0, acc, false, false);

        // ---- logits + softmax over 16 classes ----
        float lg[8];
        #pragma unroll
        for (int r = 0; r < 8; ++r) lg[r] = acc[r] + bcr[r];

        float mx = lg[0];
        #pragma unroll
        for (int r = 1; r < 8; ++r) mx = fmaxf(mx, lg[r]);
        mx = fmaxf(mx, swap16(mx));              // combine both class halves

        float ex[8];
        float s = 0.0f;
        #pragma unroll
        for (int r = 0; r < 8; ++r) { ex[r] = __expf(lg[r] - mx); s += ex[r]; }
        s += swap16(s);

        const float rs = __builtin_amdgcn_rcpf(s);

        // ---- store: 8 consecutive floats at c = 8h..8h+7 for pair i ----
        const int i = i0 + m;
        float* op = out + ((((size_t)b * NN + i) * NN) + j) * CC + 8 * h;
        float4 o0 = make_float4(ex[0] * rs, ex[1] * rs, ex[2] * rs, ex[3] * rs);
        float4 o1 = make_float4(ex[4] * rs, ex[5] * rs, ex[6] * rs, ex[7] * rs);
        ((float4*)op)[0] = o0;
        ((float4*)op)[1] = o1;
    }
}

// ---------------------------------------------------------------------------
// Launch
// ---------------------------------------------------------------------------
extern "C" void kernel_launch(void* const* d_in, const int* in_sizes, int n_in,
                              void* d_out, int out_size, void* d_ws, size_t ws_size,
                              hipStream_t stream) {
    (void)in_sizes; (void)n_in; (void)out_size; (void)ws_size;

    const float* features = (const float*)d_in[0];   // [B,N,D]
    const float* Wk       = (const float*)d_in[1];   // [D,D]
    const float* bk       = (const float*)d_in[2];   // [D]
    const float* Wq       = (const float*)d_in[3];   // [D,D]
    const float* bq       = (const float*)d_in[4];   // [D]
    const float* Wc       = (const float*)d_in[5];   // [C,D]
    const float* bc       = (const float*)d_in[6];   // [C]
    float*       out      = (float*)d_out;           // [B,N,N,C]

    float* key2   = (float*)d_ws;                    // B*N*D floats
    float* query2 = key2 + (size_t)BB * NN * DD;     // B*N*D floats

    ncl_linear2_kernel<<<dim3(BB * NN), dim3(DD), 0, stream>>>(
        features, Wk, bk, Wq, bq, key2, query2);

    // 4096 tiles = 512 blocks x 8 waves (wave32, blockDim 256)
    ncl_pair_softmax_kernel<<<dim3(512), dim3(256), 0, stream>>>(
        key2, query2, Wc, bc, out);
}
